// GAT_12695923327311
// MI455X (gfx1250) — matrix operand order repaired and checked by reading, not compiled, not run on hardware
//
#include <hip/hip_runtime.h>
#include <hip/hip_bf16.h>

typedef __attribute__((ext_vector_type(16))) _Float16 v16h;
typedef __attribute__((ext_vector_type(8)))  _Float16 v8h;
typedef __attribute__((ext_vector_type(8)))  float    v8f;

#define F_IN 256
#define HID 64
#define HEADS 2
#define NCLS 40
#define NC2PAD 48
#define NEG_SLOPE 0.2f

union V16 { v16h v; v8h h[2]; };

// ---------------- fp32 -> (hi,lo) fp16 split ----------------
__device__ __forceinline__ void split32(float x, _Float16& hi, _Float16& lo) {
  _Float16 h = (_Float16)x;
  hi = h;
  lo = (_Float16)(x - (float)h);
}

__global__ void k_split(const float* __restrict__ in, _Float16* __restrict__ hi,
                        _Float16* __restrict__ lo, long n) {
  long i = (long)blockIdx.x * blockDim.x + threadIdx.x;
  if (i >= n) return;
  split32(in[i], hi[i], lo[i]);
}

// W[k][Ncols] row-major -> Bt[n][K] (transposed, zero-padded to NcolsPad), hi/lo split
__global__ void k_transpose_split(const float* __restrict__ W, int K, int Ncols, int NcolsPad,
                                  _Float16* __restrict__ Bth, _Float16* __restrict__ Btl) {
  int i = blockIdx.x * blockDim.x + threadIdx.x;
  if (i >= NcolsPad * K) return;
  int n = i / K, k = i % K;
  float x = (n < Ncols) ? W[(long)k * Ncols + n] : 0.0f;
  split32(x, Bth[i], Btl[i]);
}

// ---------------- WMMA GEMM: C[M,ldc] = A[M,K] * Bt[n][K]^T ----------------
// One wave per 16x16 tile. EXEC stays all-ones (WMMA requirement): rows clamped
// for loads, stores guarded per element.
__global__ void k_gemm_wmma(const _Float16* __restrict__ Ah, const _Float16* __restrict__ Al,
                            const _Float16* __restrict__ Bth, const _Float16* __restrict__ Btl,
                            float* __restrict__ C, int K, int ldc, int M) {
  const int mt   = blockIdx.x;
  const int nt   = blockIdx.y;
  const int lane = threadIdx.x;        // 0..31
  const int hh   = lane >> 4;          // lane half
  const int lr   = lane & 15;

  int row = mt * 16 + lr;
  if (row >= M) row = M - 1;           // clamp, keep EXEC all-ones
  const int col = nt * 16 + lr;

  const _Float16* arh = Ah  + (long)row * K + 16 * hh;
  const _Float16* arl = Al  + (long)row * K + 16 * hh;
  const _Float16* brh = Bth + (long)col * K + 16 * hh;
  const _Float16* brl = Btl + (long)col * K + 16 * hh;

  v8f c = {};
  for (int k0 = 0; k0 < K; k0 += 32) {
    V16 ah, al, bh, bl;
    // A 16x32 f16 fragment: lanes 0-15 rows M, K = 16h .. 16h+15 contiguous
    ah.h[0] = *(const v8h*)(arh + k0);
    ah.h[1] = *(const v8h*)(arh + k0 + 8);
    al.h[0] = *(const v8h*)(arl + k0);
    al.h[1] = *(const v8h*)(arl + k0 + 8);
    // B 32x16 f16 fragment: col = lane&15, K = k0+16h .. +16 contiguous (pre-transposed W)
    bh.v = *(const v16h*)(brh + k0);
    bl.v = *(const v16h*)(brl + k0);
    c = __builtin_amdgcn_wmma_f32_16x16x32_f16(false, ah.v, false, bh.v, (short)0, c, false, false);
    c = __builtin_amdgcn_wmma_f32_16x16x32_f16(false, ah.v, false, bl.v, (short)0, c, false, false);
    c = __builtin_amdgcn_wmma_f32_16x16x32_f16(false, al.v, false, bh.v, (short)0, c, false, false);
  }

  // C layout: VGPR r -> row = r + 8*hh, col = lane&15
  const int rbase = mt * 16 + 8 * hh;
#pragma unroll
  for (int r = 0; r < 8; ++r) {
    int rr = rbase + r;
    if (rr < M) C[(long)rr * ldc + nt * 16 + lr] = c[r];
  }
}

// ---------------- attention scores a_s[n,h], a_d[n,h] ----------------
__global__ void k_attn(const float* __restrict__ h, int ldh, int H, int Cc,
                       const float* __restrict__ att_s, const float* __restrict__ att_d,
                       float* __restrict__ as_, float* __restrict__ ad_, int N) {
  long t = (long)blockIdx.x * blockDim.x + threadIdx.x;
  if (t >= (long)N * H) return;
  long n = t / H; int hd = (int)(t % H);
  const float* hp = h + n * ldh + hd * Cc;
  const float* sp = att_s + hd * Cc;
  const float* dp = att_d + hd * Cc;
  float s = 0.f, d = 0.f;
  for (int c = 0; c < Cc; ++c) { float v = hp[c]; s += v * sp[c]; d += v * dp[c]; }
  as_[t] = s; ad_[t] = d;
}

// ---------------- edge pass 1: ex = exp(leakyrelu(a_s[src]+a_d[dst])); denom[dst] += ex
__global__ void k_edge_softmax_num(const int* __restrict__ ei, long E, long Etot, int H,
                                   const float* __restrict__ as_, const float* __restrict__ ad_,
                                   float* __restrict__ ex, float* __restrict__ denom) {
  long t = (long)blockIdx.x * blockDim.x + threadIdx.x;
  if (t >= Etot * H) return;
  long e = t / H; int hd = (int)(t % H);
  int s, d;
  if (e < E) { s = ei[e]; d = ei[E + e]; } else { s = d = (int)(e - E); }
  float v = as_[(long)s * H + hd] + ad_[(long)d * H + hd];
  v = (v > 0.f) ? v : NEG_SLOPE * v;       // softmax shift-invariant: max-subtract omitted
  float xv = __expf(v);
  ex[t] = xv;
  atomicAdd(&denom[(long)d * H + hd], xv);
}

// ---------------- edge pass 2: out[dst] += h[src] * alpha (one wave32 per edge)
__global__ void k_edge_aggregate(const int* __restrict__ ei, long E, long Etot, int H, int Cc,
                                 const float* __restrict__ hfeat, int ldh,
                                 const float* __restrict__ ex, const float* __restrict__ denom,
                                 float* __restrict__ out, int ldo) {
  long t = (long)blockIdx.x * blockDim.x + threadIdx.x;
  long e = t >> 5;
  int lane = (int)(t & 31);
  if (e >= Etot) return;
  int s, d;
  if (e < E) { s = ei[e]; d = ei[E + e]; } else { s = d = (int)(e - E); }
  const int HC = H * Cc;
  for (int j = lane; j < HC; j += 32) {
    int hd = j / Cc;
    float a = ex[e * H + hd] / (denom[(long)d * H + hd] + 1e-16f);
    atomicAdd(&out[(long)d * ldo + j], hfeat[(long)s * ldh + j] * a);
  }
}

// ---------------- relu(x + bias) then hi/lo f16 split (feeds layer-2 GEMM) ----
__global__ void k_relu_split(const float* __restrict__ in, const float* __restrict__ bias, int ld,
                             _Float16* __restrict__ hi, _Float16* __restrict__ lo, long n) {
  long i = (long)blockIdx.x * blockDim.x + threadIdx.x;
  if (i >= n) return;
  float v = in[i] + bias[i % ld];
  v = v > 0.f ? v : 0.f;
  split32(v, hi[i], lo[i]);
}

// ---------------- log_softmax over NCLS classes ----------------
__global__ void k_logsoftmax(const float* __restrict__ in, const float* __restrict__ bias,
                             float* __restrict__ out, int N) {
  long n = (long)blockIdx.x * blockDim.x + threadIdx.x;
  if (n >= N) return;
  float v[NCLS];
  float m = -1e30f;
  for (int c = 0; c < NCLS; ++c) { v[c] = in[n * NCLS + c] + bias[c]; m = fmaxf(m, v[c]); }
  float s = 0.f;
  for (int c = 0; c < NCLS; ++c) s += __expf(v[c] - m);
  float ls = __logf(s);
  for (int c = 0; c < NCLS; ++c) out[n * NCLS + c] = v[c] - m - ls;
}

static inline long cdiv(long a, long b) { return (a + b - 1) / b; }

extern "C" void kernel_launch(void* const* d_in, const int* in_sizes, int n_in,
                              void* d_out, int out_size, void* d_ws, size_t ws_size,
                              hipStream_t stream) {
  const float* x      = (const float*)d_in[0];
  const int*   ei     = (const int*)d_in[1];
  const float* W1     = (const float*)d_in[2];
  const float* att_s1 = (const float*)d_in[3];
  const float* att_d1 = (const float*)d_in[4];
  const float* b1     = (const float*)d_in[5];
  const float* W2     = (const float*)d_in[6];
  const float* att_s2 = (const float*)d_in[7];
  const float* att_d2 = (const float*)d_in[8];
  const float* b2     = (const float*)d_in[9];
  float* out = (float*)d_out;

  const long N    = in_sizes[0] / F_IN;
  const long E    = in_sizes[1] / 2;
  const long Etot = E + N;
  const int  HC1  = HEADS * HID;  // 128

  // ---- workspace carve (regions reused across phases) ----
  char* p = (char*)d_ws;
  auto carve = [&](size_t bytes) { void* r = (void*)p; p += (bytes + 255) & ~(size_t)255; return r; };

  // Region A/B: x f16 split (layer1 A); reused as relu'd h f16 split (layer2 A).
  _Float16* xh = (_Float16*)carve(N * F_IN * sizeof(_Float16));   // later: oh (N*128)
  _Float16* xl = (_Float16*)carve(N * F_IN * sizeof(_Float16));   // later: ol
  // Region C: h1 (N x 128); reused as out2 (N x 40) after layer-1 aggregation.
  float* h1 = (float*)carve(N * HC1 * sizeof(float));
  // Region D: out1 (N x 128); reused as h2 (N x 48) after relu_split.
  float* out1 = (float*)carve(N * HC1 * sizeof(float));
  // Region E: per-edge exp, layer1 (Etot x 2); reused for layer2 (Etot x 1).
  float* ex1 = (float*)carve(Etot * HEADS * sizeof(float));
  // Small persistent buffers.
  _Float16* W1th = (_Float16*)carve((size_t)HC1 * F_IN * sizeof(_Float16));
  _Float16* W1tl = (_Float16*)carve((size_t)HC1 * F_IN * sizeof(_Float16));
  _Float16* W2th = (_Float16*)carve((size_t)NC2PAD * HC1 * sizeof(_Float16));
  _Float16* W2tl = (_Float16*)carve((size_t)NC2PAD * HC1 * sizeof(_Float16));
  float* as1    = (float*)carve(N * HEADS * sizeof(float));
  float* ad1    = (float*)carve(N * HEADS * sizeof(float));
  float* denom1 = (float*)carve(N * HEADS * sizeof(float));
  float* as2    = (float*)carve(N * sizeof(float));
  float* ad2    = (float*)carve(N * sizeof(float));
  float* denom2 = (float*)carve(N * sizeof(float));
  // Aliases
  _Float16* oh = xh;          // N*128 f16
  _Float16* ol = xl;
  float* out2  = h1;          // N*40
  float* h2    = out1;        // N*48
  float* ex2   = ex1;         // Etot*1

  const long mtiles = cdiv(N, 16);

  // ---- phase 0: conversions ----
  k_split<<<cdiv(N * F_IN, 256), 256, 0, stream>>>(x, xh, xl, N * F_IN);
  k_transpose_split<<<cdiv((long)HC1 * F_IN, 256), 256, 0, stream>>>(W1, F_IN, HC1, HC1, W1th, W1tl);
  k_transpose_split<<<cdiv((long)NC2PAD * HC1, 256), 256, 0, stream>>>(W2, HC1, NCLS, NC2PAD, W2th, W2tl);
  hipMemsetAsync(denom1, 0, N * HEADS * sizeof(float), stream);
  hipMemsetAsync(out1, 0, N * HC1 * sizeof(float), stream);

  // ---- layer 1 ----
  k_gemm_wmma<<<dim3((unsigned)mtiles, HC1 / 16), 32, 0, stream>>>(
      xh, xl, W1th, W1tl, h1, F_IN, HC1, (int)N);
  k_attn<<<cdiv(N * HEADS, 256), 256, 0, stream>>>(h1, HC1, HEADS, HID, att_s1, att_d1, as1, ad1, (int)N);
  k_edge_softmax_num<<<cdiv(Etot * HEADS, 256), 256, 0, stream>>>(ei, E, Etot, HEADS, as1, ad1, ex1, denom1);
  k_edge_aggregate<<<cdiv(Etot * 32, 256), 256, 0, stream>>>(
      ei, E, Etot, HEADS, HID, h1, HC1, ex1, denom1, out1, HC1);
  k_relu_split<<<cdiv(N * HC1, 256), 256, 0, stream>>>(out1, b1, HC1, oh, ol, N * HC1);

  // h1 / out1 / ex1 regions now dead -> safe to reuse (memsets ordered after last reads)
  hipMemsetAsync(denom2, 0, N * sizeof(float), stream);
  hipMemsetAsync(out2, 0, N * NCLS * sizeof(float), stream);

  // ---- layer 2 ----
  k_gemm_wmma<<<dim3((unsigned)mtiles, NC2PAD / 16), 32, 0, stream>>>(
      oh, ol, W2th, W2tl, h2, HC1, NC2PAD, (int)N);
  k_attn<<<cdiv(N, 256), 256, 0, stream>>>(h2, NC2PAD, 1, NCLS, att_s2, att_d2, as2, ad2, (int)N);
  k_edge_softmax_num<<<cdiv(Etot, 256), 256, 0, stream>>>(ei, E, Etot, 1, as2, ad2, ex2, denom2);
  k_edge_aggregate<<<cdiv(Etot * 32, 256), 256, 0, stream>>>(
      ei, E, Etot, 1, NCLS, h2, NC2PAD, ex2, denom2, out2, NCLS);
  k_logsoftmax<<<cdiv(N, 256), 256, 0, stream>>>(out2, b2, out, (int)N);
}
